// InputTransformer_87024627352058
// MI455X (gfx1250) — matrix operand order repaired and checked by reading, not compiled, not run on hardware
//
#include <hip/hip_runtime.h>

#define D 128
#define SELU_SCALE 1.0507009873554805f
#define SELU_ALPHA 1.6732632423543772f
#define H1_STRIDE 136   // 128 + 8 bf16 pad -> 272B row pitch, kills LDS bank conflicts, 16B aligned

typedef __attribute__((ext_vector_type(16))) __bf16 v16bf;
typedef __attribute__((ext_vector_type(8)))  __bf16 v8bf;
typedef __attribute__((ext_vector_type(8)))  float  v8f;

// ---------------------------------------------------------------------------
// 1) agg = x   (residual folded into the accumulator init; also re-inits the
//    workspace deterministically every call)
// ---------------------------------------------------------------------------
__global__ void init_agg_kernel(const float4* __restrict__ x4,
                                float4* __restrict__ agg4, int n4) {
    int i = blockIdx.x * blockDim.x + threadIdx.x;
    if (i < n4) agg4[i] = x4[i];
}

// ---------------------------------------------------------------------------
// 2) W1,W2 (128x128 f32 row-major, out = x @ W) -> bf16 transposed (col-major)
//    so a WMMA B-fragment is 16 contiguous bf16 per lane.
// ---------------------------------------------------------------------------
__global__ void prep_weights_kernel(const float* __restrict__ W1,
                                    const float* __restrict__ W2,
                                    __bf16* __restrict__ w1t,
                                    __bf16* __restrict__ w2t) {
    int idx = blockIdx.x * blockDim.x + threadIdx.x;   // 0 .. 16383
    int k = idx >> 7;       // row of W  (input dim)
    int n = idx & 127;      // col of W  (output dim)
    w1t[n * D + k] = (__bf16)W1[idx];
    w2t[n * D + k] = (__bf16)W2[idx];
}

// ---------------------------------------------------------------------------
// 3) Edge scatter: one wave32 per edge. Lane l owns feature bytes [l*16, l*16+16):
//    coalesced float4 gather of x[src] (L2-resident, 51.2MB < 192MB L2),
//    4x global_atomic_add_f32 into agg[dst] (also L2-resident).
// ---------------------------------------------------------------------------
__global__ void scatter_kernel(const float4* __restrict__ x4,
                               const int* __restrict__ esrc,
                               const int* __restrict__ edst,
                               const float* __restrict__ ew,
                               float* __restrict__ agg, int E) {
    int wid  = threadIdx.x >> 5;
    int lane = threadIdx.x & 31;
    int e = blockIdx.x * (blockDim.x >> 5) + wid;
    if (e >= E) return;
    int src = esrc[e];
    int dst = edst[e];
    float w = ew[e];
    float4 v = x4[src * (D / 4) + lane];
    float* o = agg + dst * D + lane * 4;
    atomicAdd(o + 0, v.x * w);
    atomicAdd(o + 1, v.y * w);
    atomicAdd(o + 2, v.z * w);
    atomicAdd(o + 3, v.w * w);
}

__device__ __forceinline__ float selu_f(float v) {
    return v > 0.f ? SELU_SCALE * v
                   : (SELU_SCALE * SELU_ALPHA) * (__expf(v) - 1.f);
}

// ---------------------------------------------------------------------------
// 4) Fused 2-layer MLP with SELU, bf16 WMMA / f32 accumulate.
//    Block = 256 threads = 8 waves; block owns 16 rows; wave w owns the
//    16-column tile [w*16, w*16+16). Layer-1 output staged in LDS as bf16.
//
//    ISA fragment layouts (wave32, 16x16x32 bf16):
//      A: lane L -> row M = L%16, elems 0..7:  K = (L/16)*8 + 0..7
//                                 elems 8..15: K = 16 + (L/16)*8 + 0..7
//      B: lane L -> col N = L%16, same K order as A (this code feeds A and B with K = (L/16)*16 + 0..15)
//      C/D: vgpr g, lane L -> M = g + 8*(L/16), N = L%16
// ---------------------------------------------------------------------------
__global__ void __launch_bounds__(256)
mlp_kernel(const float* __restrict__ agg,
           const __bf16* __restrict__ w1t,
           const float* __restrict__ b1,
           const __bf16* __restrict__ w2t,
           const float* __restrict__ b2,
           float* __restrict__ out) {
    __shared__ __align__(16) __bf16 h1[16 * H1_STRIDE];

    const int lane = threadIdx.x & 31;
    const int wid  = threadIdx.x >> 5;   // column tile 0..7
    const int half = lane >> 4;          // 0 or 1
    const int mrow = lane & 15;
    const int rowbase = blockIdx.x * 16;
    const int ncol = wid * 16 + mrow;    // this lane's output column

    // ---------------- Layer 1: C = bf16(agg_rows) x W1T ----------------
    v8f c1 = {};
#pragma unroll
    for (int kb = 0; kb < D; kb += 32) {
        // A fragment: row (rowbase+mrow), K = kb + half*16 + 0..15
        const float* arow = agg + (size_t)(rowbase + mrow) * D + kb + half * 16;
        float4 f0 = *(const float4*)(arow);
        float4 f1 = *(const float4*)(arow + 4);
        float4 f2 = *(const float4*)(arow + 8);
        float4 f3 = *(const float4*)(arow + 12);
        v16bf a;
        a[0]  = (__bf16)f0.x; a[1]  = (__bf16)f0.y; a[2]  = (__bf16)f0.z; a[3]  = (__bf16)f0.w;
        a[4]  = (__bf16)f1.x; a[5]  = (__bf16)f1.y; a[6]  = (__bf16)f1.z; a[7]  = (__bf16)f1.w;
        a[8]  = (__bf16)f2.x; a[9]  = (__bf16)f2.y; a[10] = (__bf16)f2.z; a[11] = (__bf16)f2.w;
        a[12] = (__bf16)f3.x; a[13] = (__bf16)f3.y; a[14] = (__bf16)f3.z; a[15] = (__bf16)f3.w;
        // B fragment: col ncol, 16 contiguous bf16 starting at kb + half*16
        v16bf b = *(const v16bf*)(w1t + ncol * D + kb + half * 16);
        c1 = __builtin_amdgcn_wmma_f32_16x16x32_bf16(false, a, false, b,
                                                     (short)0, c1, false, false);
    }
    {
        float bias1 = b1[ncol];
#pragma unroll
        for (int g = 0; g < 8; ++g) {
            float v = selu_f(c1[g] + bias1);
            h1[(g + 8 * half) * H1_STRIDE + ncol] = (__bf16)v;
        }
    }
    __syncthreads();

    // ---------------- Layer 2: C = h1(LDS bf16) x W2T ----------------
    v8f c2 = {};
#pragma unroll
    for (int kb = 0; kb < D; kb += 32) {
        const __bf16* hrow = h1 + mrow * H1_STRIDE + kb + half * 16;
        v8bf lo = *(const v8bf*)(hrow);        // K = kb + half*16 + 0..7
        v8bf hi = *(const v8bf*)(hrow + 8);    // K = kb + half*16 + 8..15
        v16bf a = __builtin_shufflevector(lo, hi,
                   0, 1, 2, 3, 4, 5, 6, 7, 8, 9, 10, 11, 12, 13, 14, 15);
        v16bf b = *(const v16bf*)(w2t + ncol * D + kb + half * 16);
        c2 = __builtin_amdgcn_wmma_f32_16x16x32_bf16(false, a, false, b,
                                                     (short)0, c2, false, false);
    }
    {
        float bias2 = b2[ncol];
#pragma unroll
        for (int g = 0; g < 8; ++g) {
            float v = selu_f(c2[g] + bias2);
            out[(size_t)(rowbase + g + 8 * half) * D + ncol] = v;
        }
    }
}

// ---------------------------------------------------------------------------
// Launcher
// ---------------------------------------------------------------------------
extern "C" void kernel_launch(void* const* d_in, const int* in_sizes, int n_in,
                              void* d_out, int out_size, void* d_ws, size_t ws_size,
                              hipStream_t stream) {
    const float* x    = (const float*)d_in[0];
    const int*   esrc = (const int*)  d_in[1];
    const int*   edst = (const int*)  d_in[2];
    const float* ew   = (const float*)d_in[3];
    const float* W1   = (const float*)d_in[4];
    const float* b1   = (const float*)d_in[5];
    const float* W2   = (const float*)d_in[6];
    const float* b2   = (const float*)d_in[7];

    const int ND = in_sizes[0];     // N * 128
    const int N  = ND / D;
    const int E  = in_sizes[1];

    // Workspace layout: [ agg f32 N*D | w1t bf16 128*128 | w2t bf16 128*128 ]
    float*  agg = (float*)d_ws;
    __bf16* w1t = (__bf16*)((char*)d_ws + (size_t)ND * sizeof(float));
    __bf16* w2t = w1t + D * D;
    float*  out = (float*)d_out;

    // 1) agg = x (residual pre-folded)
    int n4 = ND / 4;
    init_agg_kernel<<<(n4 + 255) / 256, 256, 0, stream>>>(
        (const float4*)x, (float4*)agg, n4);

    // 2) weights -> transposed bf16
    prep_weights_kernel<<<(D * D) / 256, 256, 0, stream>>>(W1, W2, w1t, w2t);

    // 3) edge scatter: wave per edge, 8 edges per 256-thread block
    scatter_kernel<<<(E + 7) / 8, 256, 0, stream>>>(
        (const float4*)x, esrc, edst, ew, agg, E);

    // 4) fused 2-layer MLP (N divisible by 16: 100000 = 6250 * 16)
    mlp_kernel<<<N / 16, 256, 0, stream>>>(agg, w1t, b1, w2t, b2, out);
}
